// SelfAttention_51548197487426
// MI455X (gfx1250) — matrix operand order repaired and checked by reading, not compiled, not run on hardware
//
#include <hip/hip_runtime.h>
#include <hip/hip_bf16.h>
#include <math.h>

typedef __bf16 bf16_t;
typedef __bf16 v16bf __attribute__((ext_vector_type(16)));
typedef __bf16 v8bf  __attribute__((ext_vector_type(8)));
typedef float  v8f   __attribute__((ext_vector_type(8)));
typedef float  v4f   __attribute__((ext_vector_type(4)));

#define B_ 16
#define L_ 512
#define H_ 768
#define C_ 4096

#define SC_STRIDE 516   // floats per score row (512 + 4 pad: conflict-free D stores)
#define AT_STRIDE 520   // bf16 per attn row (512 + 8 pad)
#define WL_STRIDE 776   // bf16 per W row   (768 + 8 pad)
#define TL_STRIDE 40    // bf16 per T row   (32 l + 8 pad: stride 20 words, conflict-free)

#define SC_BYTES (16 * SC_STRIDE * 4)
#define AT_BYTES (16 * AT_STRIDE * 2)
#define WL_BYTES (16 * WL_STRIDE * 2)
#define TL_BYTES (H_ * TL_STRIDE * 2)
#define SMEM_BYTES (SC_BYTES + AT_BYTES + WL_BYTES + TL_BYTES)   // 135,936 B < 160 KB

// f32 -> bf16. Plain fptrunc: clang lowers to the native packed conversion
// (v_cvt_pk_bf16_f32-class, RNE) on gfx1250 instead of a 3-VALU bit sequence.
__device__ __forceinline__ bf16_t f2bf(float f) {
  return static_cast<bf16_t>(f);
}

__device__ __forceinline__ v16bf bf_cat(v8bf lo, v8bf hi8) {
  v16bf r;
#pragma unroll
  for (int i = 0; i < 8; ++i) { r[i] = lo[i]; r[i + 8] = hi8[i]; }
  return r;
}

__device__ __forceinline__ v16bf bf_cvt16(v4f a, v4f b, v4f c, v4f d) {
  v16bf r;
#pragma unroll
  for (int i = 0; i < 4; ++i) {
    r[i]      = f2bf(a[i]);
    r[4 + i]  = f2bf(b[i]);
    r[8 + i]  = f2bf(c[i]);
    r[12 + i] = f2bf(d[i]);
  }
  return r;
}

// Fused label-attention: scores = X W^T (masked softmax over L), out = attn X.
// One block = one (batch, 16-label) tile; 4 waves cooperate.
__global__ void __launch_bounds__(128)
label_attn_fused_kernel(const float* __restrict__ X,          // [B, L, H] f32
                        const unsigned char* __restrict__ Mk, // [B, 1, L] bool
                        const float* __restrict__ W,          // [C, H] f32
                        float* __restrict__ O) {              // [B, C, H] f32
  extern __shared__ char smem[];
  float*  sc = (float*)smem;                                     // scores f32 [16][516]
  bf16_t* at = (bf16_t*)(smem + SC_BYTES);                       // attn  bf16 [16][520]
  bf16_t* wl = (bf16_t*)(smem + SC_BYTES + AT_BYTES);            // W     bf16 [16][776]
  bf16_t* tl = (bf16_t*)(smem + SC_BYTES + AT_BYTES + WL_BYTES); // X^T   bf16 [768][40]

  const int tid  = threadIdx.x;
  const int w    = tid >> 5;        // wave 0..3
  const int lane = tid & 31;
  const int nl   = lane & 15;       // row/col within tile
  const int hi   = lane >> 4;       // half-wave select
  const int b    = blockIdx.x / (C_ / 16);
  const int c0   = (blockIdx.x % (C_ / 16)) * 16;

  // ---- Stage W[c0:c0+16, :] as bf16 into LDS (coalesced over h) ----
  for (int idx = tid; idx < 16 * H_; idx += 128) {
    int row = idx / H_;
    int h   = idx - row * H_;
    wl[row * WL_STRIDE + h] = f2bf(W[(size_t)(c0 + row) * H_ + h]);
  }
  __syncthreads();

  // ---- Phase 1: scores[c, l] = sum_h X[b,l,h] * W[c,h]  (K = H = 24 x 32) ----
  // Wave w owns L-tiles {w, 4+w, ..., 28+w}.
  for (int lt = 0; lt < 8; ++lt) {
    const int l0 = (lt * 4 + w) * 16;
    v8f acc = {0.f, 0.f, 0.f, 0.f, 0.f, 0.f, 0.f, 0.f};
    const float* xrow = X + ((size_t)b * L_ + (l0 + nl)) * H_;
#pragma unroll
    for (int kc = 0; kc < 24; ++kc) {
      const int k0 = kc * 32;
      // A fragment (W rows): lane row = nl, K = k0 + hi*16 + j (contiguous)
      v8bf alo = *(const v8bf*)(wl + nl * WL_STRIDE + k0 + hi * 16);
      v8bf ahi = *(const v8bf*)(wl + nl * WL_STRIDE + k0 + hi * 16 + 8);
      v16bf a  = bf_cat(alo, ahi);
      // B fragment (X): lane col = l0+nl, K = k0 + hi*16 + j (contiguous 64 B)
      const v4f* xp = (const v4f*)(xrow + k0 + hi * 16);
      v16bf bm = bf_cvt16(xp[0], xp[1], xp[2], xp[3]);
      acc = __builtin_amdgcn_wmma_f32_16x16x32_bf16(false, a, false, bm,
                                                    (short)0, acc, false, false);
    }
    // D layout: element (m, n) = (r + hi*8, nl)
#pragma unroll
    for (int r = 0; r < 8; ++r)
      sc[(r + hi * 8) * SC_STRIDE + l0 + nl] = acc[r];
  }
  __syncthreads();

  // ---- Phase 2: masked softmax over L, per row, 32 lanes per row (f32) ----
  for (int rr = 0; rr < 4; ++rr) {
    const int row = rr * 4 + w;
    float v[16];
    float mx = -__builtin_inff();
#pragma unroll
    for (int i = 0; i < 16; ++i) {
      const int l = lane + 32 * i;
      float s = sc[row * SC_STRIDE + l];
      s = Mk[b * L_ + l] ? s : -__builtin_inff();
      v[i] = s;
      mx = fmaxf(mx, s);
    }
#pragma unroll
    for (int off = 16; off >= 1; off >>= 1)
      mx = fmaxf(mx, __shfl_xor(mx, off, 32));
    float sum = 0.f;
#pragma unroll
    for (int i = 0; i < 16; ++i) { v[i] = __expf(v[i] - mx); sum += v[i]; }
#pragma unroll
    for (int off = 16; off >= 1; off >>= 1)
      sum += __shfl_xor(sum, off, 32);
    const float inv = 1.0f / sum;
#pragma unroll
    for (int i = 0; i < 16; ++i)
      at[row * AT_STRIDE + lane + 32 * i] = f2bf(v[i] * inv);
  }
  __syncthreads();

  // ---- Phase 3: out[c, h] = sum_l attn[c,l] * X[b,l,h]  (K = L = 16 x 32) ----
  // Wave w owns H-tiles {w, 4+w, ..., 44+w}: 12 persistent accumulators.
  v8f accO[12];
#pragma unroll
  for (int nt = 0; nt < 12; ++nt)
#pragma unroll
    for (int r = 0; r < 8; ++r) accO[nt][r] = 0.f;

  for (int lc = 0; lc < 16; ++lc) {
    const int l0 = lc * 32;
    // Stage X[b, l0:l0+32, :] transposed to tl[h][l] as bf16 (coalesced over h),
    // and prefetch the next chunk into cache (global_prefetch_b8) to overlap
    // HBM latency with this chunk's WMMA work.
    const size_t xbase = (size_t)b * L_ * H_;
    for (int idx = tid; idx < 32 * H_; idx += 128) {
      int l = idx / H_;
      int h = idx - l * H_;
      tl[h * TL_STRIDE + l] = f2bf(X[xbase + (size_t)(l0 + l) * H_ + h]);
      if (lc < 15)
        __builtin_prefetch(&X[xbase + (size_t)(l0 + 32 + l) * H_ + h], 0, 0);
    }
    __syncthreads();

    // A fragment (attn rows) — identical for every n-tile this chunk
    const int k0 = lc * 32;
    v8bf alo = *(const v8bf*)(at + nl * AT_STRIDE + k0 + hi * 16);
    v8bf ahi = *(const v8bf*)(at + nl * AT_STRIDE + k0 + hi * 16 + 8);
    v16bf a  = bf_cat(alo, ahi);

#pragma unroll
    for (int nt = 0; nt < 12; ++nt) {
      const int h0 = (nt * 4 + w) * 16 + nl;   // lane's column (absolute h)
      // B fragment: lane col = h0, K = hi*16 + j within chunk (contiguous 32 B)
      v8bf b0 = *(const v8bf*)(tl + h0 * TL_STRIDE + hi * 16);
      v8bf b1 = *(const v8bf*)(tl + h0 * TL_STRIDE + hi * 16 + 8);
      v16bf bm = bf_cat(b0, b1);
      accO[nt] = __builtin_amdgcn_wmma_f32_16x16x32_bf16(false, a, false, bm,
                                                         (short)0, accO[nt], false, false);
    }
    __syncthreads();
  }

  // ---- Store out tile (coalesced over h across lanes) ----
#pragma unroll
  for (int nt = 0; nt < 12; ++nt) {
    const int h0 = (nt * 4 + w) * 16 + nl;
#pragma unroll
    for (int r = 0; r < 8; ++r) {
      const int m = r + hi * 8;
      O[((size_t)b * C_ + c0 + m) * H_ + h0] = accO[nt][r];
    }
  }
}

extern "C" void kernel_launch(void* const* d_in, const int* in_sizes, int n_in,
                              void* d_out, int out_size, void* d_ws, size_t ws_size,
                              hipStream_t stream) {
  (void)in_sizes; (void)n_in; (void)d_ws; (void)ws_size; (void)out_size;
  const float*         X  = (const float*)d_in[0];
  const unsigned char* Mk = (const unsigned char*)d_in[1];
  const float*         W  = (const float*)d_in[2];
  float*               O  = (float*)d_out;

  // Allow >64 KB dynamic LDS (capture-safe: not a stream op, deterministic).
  hipFuncSetAttribute((const void*)label_attn_fused_kernel,
                      hipFuncAttributeMaxDynamicSharedMemorySize, SMEM_BYTES);

  dim3 grid(B_ * (C_ / 16));   // 4096 blocks: (batch, 16-label tile)
  dim3 block(128);             // 4 waves
  label_attn_fused_kernel<<<grid, block, SMEM_BYTES, stream>>>(X, Mk, W, O);
}
